// RoutedLoRA_58634893525637
// MI455X (gfx1250) — compile-verified, matrix operand order checked by reading, not run on hardware
//
#include <hip/hip_runtime.h>
#include <hip/hip_bf16.h>

typedef __bf16 bf16;
typedef __attribute__((ext_vector_type(16))) __bf16 v16bf;
typedef __attribute__((ext_vector_type(8)))  __bf16 v8bf;
typedef __attribute__((ext_vector_type(8)))  float  v8f;
typedef __attribute__((ext_vector_type(2)))  float  v2f;

#define T_TOK   16384
#define IN_F    4096
#define OUT_F   4096
#define RANK    8
#define NEXP    64
#define TOPK    16
#define RDIM    16
#define BOT     512        // NEXP*RANK
#define SCALING 2.0f       // ALPHA/TOP_K = 32/16

union V16U { v16bf v; v8bf h[2]; };

static __device__ __forceinline__ v8f wmma_bf16(const V16U& a, const V16U& b, v8f c) {
  return __builtin_amdgcn_wmma_f32_16x16x32_bf16(false, a.v, false, b.v,
                                                 (short)0, c, false, false);
}
static __device__ __forceinline__ v8f wmma_f32x4(v2f a, v2f b, v8f c) {
  return __builtin_amdgcn_wmma_f32_16x16x4_f32(false, a, false, b,
                                               (short)0, c, false, false);
}

// Async global->LDS copy of 16 bytes per lane (GLOBAL_LOAD_ASYNC_TO_LDS_B128,
// GVS mode: mem = SADDR + VADDR + inst_offset; LDS dest = VDST vgpr).
static __device__ __forceinline__ void async_copy_b128(unsigned lds_off,
                                                       const void* sbase,
                                                       unsigned voff) {
  asm volatile("global_load_async_to_lds_b128 %0, %1, %2"
               :: "v"(lds_off), "v"(voff), "s"(sbase) : "memory");
}
static __device__ __forceinline__ void async_wait0() {
  asm volatile("s_wait_asynccnt 0x0" ::: "memory");
}
static __device__ __forceinline__ unsigned lds_offset_of(const void* p) {
  // flat-aperture rule: LDS byte offset is addr[31:0]
  return (unsigned)(unsigned long long)p;
}

// ---------------- K0: convert weights to bf16 ----------------
__global__ void k0_convert(const float* __restrict__ Aw, const float* __restrict__ Bw,
                           bf16* __restrict__ Abf, bf16* __restrict__ Bbf) {
  size_t i = (size_t)blockIdx.x * 256 + threadIdx.x;
  if (i < (size_t)BOT * IN_F) {    // A and B have identical element counts
    Abf[i] = (bf16)Aw[i];
    Bbf[i] = (bf16)Bw[i];
  }
}

// ---------------- K1: router (full f32 WMMA) + top-k softmax -> gate ----------------
__global__ void k1_router(const float* __restrict__ x, const float* __restrict__ Wq,
                          const float* __restrict__ keys, float* __restrict__ gate) {
  __shared__ __align__(16) float qs[8][16][16];   // per-wave q tile
  __shared__ __align__(16) float ss[8][16][64];   // per-wave score tile

  const int tid  = threadIdx.x;
  const int wave = tid >> 5;
  const int lane = tid & 31;
  const int t0   = blockIdx.x * 128 + wave * 16;  // 16 tokens per wave
  const int r    = lane & 15;
  const int koff = (lane >> 4) * 2;               // f32 WMMA A/B: pair at (lane/16)*2

  // q[16,16] = x_tile @ Wq^T   (K = 4096, 16x16x4 f32 WMMA, 2 independent chains)
  v8f qa0 = {0,0,0,0,0,0,0,0};
  v8f qa1 = {0,0,0,0,0,0,0,0};
  const float* xrow  = x  + (size_t)(t0 + r) * IN_F;
  const float* wqrow = Wq + (size_t)r * IN_F;
  for (int k = 0; k < IN_F; k += 8) {
    v2f a0 = *(const v2f*)(xrow  + k + koff);
    v2f b0 = *(const v2f*)(wqrow + k + koff);
    v2f a1 = *(const v2f*)(xrow  + k + 4 + koff);
    v2f b1 = *(const v2f*)(wqrow + k + 4 + koff);
    qa0 = wmma_f32x4(a0, b0, qa0);
    qa1 = wmma_f32x4(a1, b1, qa1);
  }
  v8f qa = qa0 + qa1;

  // D-layout -> LDS
  const int mbase = 8 * (lane >> 4);
#pragma unroll
  for (int v = 0; v < 8; ++v) qs[wave][mbase + v][r] = qa[v];
  __syncthreads();

  // scores[16,64] = q @ keys^T  (K = 16)
  v8f sa[4];
#pragma unroll
  for (int e = 0; e < 4; ++e) sa[e] = (v8f){0,0,0,0,0,0,0,0};
#pragma unroll
  for (int kt = 0; kt < 4; ++kt) {
    int kc = kt * 4 + koff;
    v2f a = { qs[wave][r][kc], qs[wave][r][kc + 1] };
#pragma unroll
    for (int et = 0; et < 4; ++et) {
      v2f b = *(const v2f*)(keys + (size_t)(et * 16 + r) * RDIM + kc);
      sa[et] = wmma_f32x4(a, b, sa[et]);
    }
  }
#pragma unroll
  for (int et = 0; et < 4; ++et)
#pragma unroll
    for (int v = 0; v < 8; ++v) ss[wave][mbase + v][et * 16 + r] = sa[et][v];
  __syncthreads();

  // lane-per-token top-16 + softmax
  if (lane < 16) {
    const float* srow = ss[wave][lane];
    unsigned lo = 0, hi = 0;
    float mx = -3.0e38f;
    for (int i = 0; i < TOPK; ++i) {
      float best = -3.0e38f; int bi = 0;
      for (int e = 0; e < NEXP; ++e) {
        unsigned used = (e < 32) ? ((lo >> e) & 1u) : ((hi >> (e - 32)) & 1u);
        float s = srow[e];
        if (!used && s > best) { best = s; bi = e; }
      }
      if (bi < 32) lo |= 1u << bi; else hi |= 1u << (bi - 32);
      if (i == 0) mx = best;
    }
    float sum = 0.0f;
    for (int e = 0; e < NEXP; ++e) {
      unsigned sel = (e < 32) ? ((lo >> e) & 1u) : ((hi >> (e - 32)) & 1u);
      if (sel) sum += __expf(srow[e] - mx);
    }
    float inv = 1.0f / sum;
    float* grow = gate + (size_t)(t0 + lane) * NEXP;
    for (int e = 0; e < NEXP; ++e) {
      unsigned sel = (e < 32) ? ((lo >> e) & 1u) : ((hi >> (e - 32)) & 1u);
      grow[e] = sel ? __expf(srow[e] - mx) * inv : 0.0f;
    }
  }
}

// ---------------- K2: z_gated[T,512] = (x @ A^T) * gate   (bf16 WMMA) ----------------
__global__ void k2_down(const float* __restrict__ x, const bf16* __restrict__ Abf,
                        const float* __restrict__ gate, bf16* __restrict__ zbf) {
  __shared__ __align__(16) bf16  xs[2][64][128];  // 2 x 16 KB double buffer
  __shared__ __align__(16) float gls[64][64];     // 16 KB gate rows

  const int tid  = threadIdx.x;
  const int wave = tid >> 5;
  const int lane = tid & 31;
  const int m0   = blockIdx.x * 64;
  const int mt   = wave & 3;        // m-tile 0..3
  const int nh   = wave >> 2;       // n-half 0..1 (16 n-tiles each)
  const int r    = lane & 15;
  const int kb   = (lane >> 4) * 8; // bf16 WMMA per-lane K base

  // async gate staging: pure f32 copy, overlaps the whole GEMM
  {
    const float*   gsrc = gate + (size_t)m0 * NEXP;
    const unsigned gbase = lds_offset_of(&gls[0][0]);
    for (int i = tid; i < 64 * 64 / 4; i += 256)     // 4 x B128 per thread
      async_copy_b128(gbase + (unsigned)i * 16u, gsrc, (unsigned)i * 16u);
  }

  v8f acc[16];
#pragma unroll
  for (int nt = 0; nt < 16; ++nt) acc[nt] = (v8f){0,0,0,0,0,0,0,0};

  // staging helpers (8 float4 per thread per 64x128 chunk)
  float4 stage[8];
  const int srow0 = (tid * 4) >> 7;       // rows advance by 8 per step of 256 threads
  const int scol  = (tid * 4) & 127;

  // prologue: chunk 0 -> buffer 0
#pragma unroll
  for (int j = 0; j < 8; ++j)
    stage[j] = *(const float4*)(x + (size_t)(m0 + srow0 + j * 8) * IN_F + scol);
#pragma unroll
  for (int j = 0; j < 8; ++j) {
    bf16* d = &xs[0][srow0 + j * 8][scol];
    d[0] = (bf16)stage[j].x; d[1] = (bf16)stage[j].y;
    d[2] = (bf16)stage[j].z; d[3] = (bf16)stage[j].w;
  }
  __syncthreads();

  for (int c = 0; c < 32; ++c) {
    const int kc  = c * 128;
    const int buf = c & 1;
    // issue next chunk's global loads (in flight during compute)
    if (c + 1 < 32) {
#pragma unroll
      for (int j = 0; j < 8; ++j)
        stage[j] = *(const float4*)(x + (size_t)(m0 + srow0 + j * 8) * IN_F + kc + 128 + scol);
    }
    // compute on current buffer
#pragma unroll
    for (int kk = 0; kk < 128; kk += 32) {
      V16U A;
      const bf16* ap = &xs[buf][mt * 16 + r][kk + kb];
      A.h[0] = *(const v8bf*)ap;
      A.h[1] = *(const v8bf*)(ap + 16);
#pragma unroll
      for (int nt = 0; nt < 16; ++nt) {
        V16U B;
        const bf16* bp = Abf + (size_t)((nh * 16 + nt) * 16 + r) * IN_F + kc + kk + kb;
        B.h[0] = *(const v8bf*)bp;
        B.h[1] = *(const v8bf*)(bp + 16);
        acc[nt] = wmma_bf16(A, B, acc[nt]);
      }
    }
    // convert + store next chunk into the other buffer
    if (c + 1 < 32) {
#pragma unroll
      for (int j = 0; j < 8; ++j) {
        bf16* d = &xs[buf ^ 1][srow0 + j * 8][scol];
        d[0] = (bf16)stage[j].x; d[1] = (bf16)stage[j].y;
        d[2] = (bf16)stage[j].z; d[3] = (bf16)stage[j].w;
      }
    }
    __syncthreads();
  }

  // make async gate copy visible to all waves
  async_wait0();
  __syncthreads();

  // gate + store z as bf16
  const int mbase = 8 * (lane >> 4);
#pragma unroll
  for (int nt = 0; nt < 16; ++nt) {
    int n = (nh * 16 + nt) * 16 + r;
#pragma unroll
    for (int v = 0; v < 8; ++v) {
      int m = mbase + v;
      int tok = m0 + mt * 16 + m;
      float g = gls[mt * 16 + m][n >> 3];   // expert = n/RANK
      zbf[(size_t)tok * BOT + n] = (bf16)(acc[nt][v] * g);
    }
  }
}

// ---------------- K3: out[T,4096] = z_gated @ B^T * SCALING  (bf16 WMMA) ----------------
__global__ void k3_up(const bf16* __restrict__ zbf, const bf16* __restrict__ Bbf,
                      float* __restrict__ out) {
  __shared__ __align__(16) bf16 zs[32][512];    // 32 KB

  const int tid  = threadIdx.x;
  const int wave = tid >> 5;
  const int lane = tid & 31;
  const int m0   = blockIdx.x * 32;
  const int mt   = wave & 1;        // m-tile 0..1
  const int col  = wave >> 1;       // wave column 0..3
  const int r    = lane & 15;
  const int kb   = (lane >> 4) * 8;

  // async z staging: pure bf16 copy via GLOBAL_LOAD_ASYNC_TO_LDS_B128
  {
    const bf16*    zsrc  = zbf + (size_t)m0 * BOT;
    const unsigned zbase = lds_offset_of(&zs[0][0]);
    for (int i = tid; i < 32 * 512 * 2 / 16; i += 256)   // 8 x B128 per thread
      async_copy_b128(zbase + (unsigned)i * 16u, zsrc, (unsigned)i * 16u);
  }
  async_wait0();
  __syncthreads();

  const int mbase = 8 * (lane >> 4);
  for (int nb = 0; nb < 8; ++nb) {
    v8f acc[8];
#pragma unroll
    for (int nt = 0; nt < 8; ++nt) acc[nt] = (v8f){0,0,0,0,0,0,0,0};
#pragma unroll
    for (int kk = 0; kk < BOT; kk += 32) {
      V16U A;
      const bf16* ap = &zs[mt * 16 + r][kk + kb];
      A.h[0] = *(const v8bf*)ap;
      A.h[1] = *(const v8bf*)(ap + 16);
#pragma unroll
      for (int nt = 0; nt < 8; ++nt) {
        int ntile = nb * 32 + col * 8 + nt;
        V16U B;
        const bf16* bp = Bbf + (size_t)(ntile * 16 + r) * BOT + kk + kb;
        B.h[0] = *(const v8bf*)bp;
        B.h[1] = *(const v8bf*)(bp + 16);
        acc[nt] = wmma_bf16(A, B, acc[nt]);
      }
    }
#pragma unroll
    for (int nt = 0; nt < 8; ++nt) {
      int n = (nb * 32 + col * 8 + nt) * 16 + r;
#pragma unroll
      for (int v = 0; v < 8; ++v) {
        int tok = m0 + mt * 16 + mbase + v;
        out[(size_t)tok * OUT_F + n] = acc[nt][v] * SCALING;
      }
    }
  }
}

extern "C" void kernel_launch(void* const* d_in, const int* in_sizes, int n_in,
                              void* d_out, int out_size, void* d_ws, size_t ws_size,
                              hipStream_t stream) {
  const float* x    = (const float*)d_in[0];
  const float* A_w  = (const float*)d_in[1];
  const float* B_w  = (const float*)d_in[2];
  const float* Wq_w = (const float*)d_in[3];
  const float* keys = (const float*)d_in[4];
  float* out = (float*)d_out;

  // workspace layout: Abf(4MB) | Bbf(4MB) | zbf(16MB) | gate(4MB)
  bf16*  Abf  = (bf16*)d_ws;
  bf16*  Bbf  = Abf + (size_t)BOT * IN_F;
  bf16*  zbf  = Bbf + (size_t)OUT_F * BOT;
  float* gate = (float*)(zbf + (size_t)T_TOK * BOT);

  k0_convert<<<(BOT * IN_F + 255) / 256, 256, 0, stream>>>(A_w, B_w, Abf, Bbf);
  k1_router<<<T_TOK / 128, 256, 0, stream>>>(x, Wq_w, keys, gate);
  k2_down<<<T_TOK / 64, 256, 0, stream>>>(x, Abf, gate, zbf);
  k3_up<<<T_TOK / 32, 256, 0, stream>>>(zbf, Bbf, out);
}
